// CascadeMVSNet_54082228191760
// MI455X (gfx1250) — matrix operand order repaired and checked by reading, not compiled, not run on hardware
//
#include <hip/hip_runtime.h>
#include <hip/hip_bf16.h>
#include <math.h>

// ---------------------------------------------------------------------------
// CascadeMVSNet cost-volume pipeline for gfx1250 (CDNA5, wave32, WMMA).
//   prep      : combine/invert projections, swizzle conv weights into WMMA
//               A-operand lane layout (16x32 f16, contiguous k per lane half).
//   warpvar   : build f16 variance volume [d][h][w][32ch] (channels innermost).
//   conv0     : 32->8(16pad) conv3d as implicit GEMM, 27x v_wmma_f32_16x16x32_f16
//               per 16(pos) x 16(outch) tile; weights staged in LDS per block
//               via global_load_async_to_lds_b128 (ASYNCcnt path).
//   convB     : 16pad->16pad (conv1, relu) and 16pad->1 f32 (conv2) with
//               K=32 spanning two adjacent w-positions x 16 padded channels.
//   depthconf : softmax over D, expected depth, 4-tap confidence.
// Whole working set (var 63MB f16 + h volumes) is L2-resident (192MB).
// ---------------------------------------------------------------------------

typedef __attribute__((ext_vector_type(16))) _Float16 v16h;
typedef __attribute__((ext_vector_type(8)))  float    v8f;
typedef __attribute__((ext_vector_type(4)))  float    f4raw;
typedef __attribute__((ext_vector_type(4)))  int      v4i;

typedef v4i __attribute__((address_space(1))) g_v4i;   // global
typedef v4i __attribute__((address_space(3))) l_v4i;   // LDS

union V16U { v16h v; f4raw q[2]; };
union V8H  { f4raw q; _Float16 h[8]; };

namespace {
constexpr int VV = 3, CC = 32, DD = 48, HH = 128, WW = 160;
constexpr int HW = HH * WW;
constexpr int NVOX = DD * HW;
constexpr int WTILES = WW / 16;            // 10 exact tiles
constexpr int TILES = DD * HH * WTILES;    // 61440 waves

// workspace offsets (bytes)
constexpr size_t OFF_RT  = 0;                                   // 2 views x (rot9+trans3)
constexpr size_t OFF_BF  = 256;                                 // 3 x 16 padded biases
constexpr size_t OFF_W0F = 512;                                 // 27*32*16 f16 = 27648B
constexpr size_t OFF_W1F = OFF_W0F + 27648;                     // 9*2*32*16 f16
constexpr size_t OFF_W2F = OFF_W1F + 18432;
constexpr size_t OFF_VAR = 65536;                               // NVOX*32 f16 = 63MB
constexpr size_t OFF_H1  = OFF_VAR + (size_t)NVOX * 32 * 2;     // NVOX*16 f16
constexpr size_t OFF_H2  = OFF_H1 + (size_t)NVOX * 16 * 2;      // NVOX*16 f16
constexpr size_t OFF_COST = OFF_VAR;   // alias: var dead once conv0 consumed it
}

// A-operand lane layout for V_WMMA_F32_16X16X32_F16 (matches B fills):
//   lane<16 : M=lane,    halves[0..15]->K=0..15
//   lane>=16: M=lane-16, halves[0..15]->K=16..31
__device__ __forceinline__ int a_k_of(int lane, int j) {
    return j + ((lane >= 16) ? 16 : 0);
}

// --- Cooperative global->LDS staging. Prefers gfx1250 async-to-LDS path
// (ASYNCcnt) when the toolchain exposes the builtin; otherwise load+ds_store.
__device__ __forceinline__ void wait_async0() {
#if __has_builtin(__builtin_amdgcn_s_wait_asynccnt)
    __builtin_amdgcn_s_wait_asynccnt(0);
#else
    asm volatile("s_wait_asynccnt 0x0" ::: "memory");
#endif
}

__device__ __forceinline__ void stage_to_lds(const _Float16* __restrict__ gsrc,
                                             _Float16* lds, int nhalves,
                                             int tid, int nthreads)
{
#if __has_builtin(__builtin_amdgcn_global_load_async_to_lds_b128)
    for (int i = tid * 8; i < nhalves; i += nthreads * 8) {
        __builtin_amdgcn_global_load_async_to_lds_b128(
            (g_v4i*)(gsrc + i),
            (l_v4i*)(lds + i),
            /*offset=*/0, /*cpol=*/0);
    }
    wait_async0();
#else
    for (int i = tid * 8; i < nhalves; i += nthreads * 8) {
        *reinterpret_cast<f4raw*>(lds + i) =
            *reinterpret_cast<const f4raw*>(gsrc + i);
    }
#endif
}

__global__ void prep_kernel(const float* __restrict__ proj,
                            const float* __restrict__ w0, const float* __restrict__ b0,
                            const float* __restrict__ w1, const float* __restrict__ b1,
                            const float* __restrict__ w2, const float* __restrict__ b2,
                            float* __restrict__ rt, float* __restrict__ bfmt,
                            _Float16* __restrict__ w0f, _Float16* __restrict__ w1f,
                            _Float16* __restrict__ w2f)
{
    const int tid = threadIdx.x;
    if (tid == 0) {
        // combined[v] = [[K R | K t],[0 0 0 1]]
        float comb[3][16];
        for (int v = 0; v < 3; ++v) {
            const float* E = proj + (v * 2 + 0) * 16;
            const float* K = proj + (v * 2 + 1) * 16;
            for (int i = 0; i < 3; ++i)
                for (int j = 0; j < 4; ++j)
                    comb[v][i * 4 + j] = K[i * 4 + 0] * E[0 + j] + K[i * 4 + 1] * E[4 + j] +
                                         K[i * 4 + 2] * E[8 + j];
            for (int j = 0; j < 4; ++j) comb[v][12 + j] = E[12 + j];
        }
        // invert ref (4x4 Gauss-Jordan, partial pivot)
        float a[4][8];
        for (int i = 0; i < 4; ++i)
            for (int j = 0; j < 4; ++j) {
                a[i][j] = comb[0][i * 4 + j];
                a[i][4 + j] = (i == j) ? 1.f : 0.f;
            }
        for (int col = 0; col < 4; ++col) {
            int piv = col;
            for (int r = col + 1; r < 4; ++r)
                if (fabsf(a[r][col]) > fabsf(a[piv][col])) piv = r;
            if (piv != col)
                for (int j = 0; j < 8; ++j) { float t = a[col][j]; a[col][j] = a[piv][j]; a[piv][j] = t; }
            float inv = 1.f / a[col][col];
            for (int j = 0; j < 8; ++j) a[col][j] *= inv;
            for (int r = 0; r < 4; ++r)
                if (r != col) {
                    float f = a[r][col];
                    for (int j = 0; j < 8; ++j) a[r][j] -= f * a[col][j];
                }
        }
        for (int v = 1; v < 3; ++v) {
            float P[3][4];
            for (int i = 0; i < 3; ++i)
                for (int j = 0; j < 4; ++j) {
                    float s2 = 0.f;
                    for (int k2 = 0; k2 < 4; ++k2) s2 += comb[v][i * 4 + k2] * a[k2][4 + j];
                    P[i][j] = s2;
                }
            float* o = rt + (v - 1) * 12;
            for (int i = 0; i < 3; ++i)
                for (int j = 0; j < 3; ++j) o[i * 3 + j] = P[i][j];
            for (int i = 0; i < 3; ++i) o[9 + i] = P[i][3];
        }
        for (int j = 0; j < 16; ++j) bfmt[j]      = (j < 8)  ? b0[j] : 0.f;
        for (int j = 0; j < 16; ++j) bfmt[16 + j] = (j < 8)  ? b1[j] : 0.f;
        for (int j = 0; j < 16; ++j) bfmt[32 + j] = (j == 0) ? b2[0] : 0.f;
    }

    // conv0 weights -> A operands, one per tap: M=outch(8 real), K=inch(32)
    for (int e = tid; e < 27 * 32 * 16; e += blockDim.x) {
        int j = e & 15, lane = (e >> 4) & 31, tap = e >> 9;
        int M = lane & 15;
        int K = a_k_of(lane, j);
        int kd = tap / 9, kh = (tap / 3) % 3, kw = tap % 3;
        float val = (M < 8) ? w0[(((M * 32 + K) * 3 + kd) * 3 + kh) * 3 + kw] : 0.f;
        w0f[e] = (_Float16)val;
    }
    // conv1/conv2 weights: per (kd,kh) row, pair s covers kw slots {2s, 2s+1}
    // K = kwslot*16 + padded_channel  (kw slot 3 and channels 8..15 are zero)
    for (int e = tid; e < 9 * 2 * 32 * 16; e += blockDim.x) {
        int j = e & 15, lane = (e >> 4) & 31, s = (e >> 9) & 1, r = e >> 10;
        int M = lane & 15;
        int K = a_k_of(lane, j);
        int kws = K >> 4, ci = K & 15, kw = s * 2 + kws;
        int kd = r / 3, kh = r % 3;
        float v1 = (M < 8 && ci < 8 && kw < 3) ? w1[(((M * 8 + ci) * 3 + kd) * 3 + kh) * 3 + kw] : 0.f;
        w1f[e] = (_Float16)v1;
        float v2 = (M == 0 && ci < 8 && kw < 3) ? w2[((ci * 3 + kd) * 3 + kh) * 3 + kw] : 0.f;
        w2f[e] = (_Float16)v2;
    }
}

// Homography warp + variance; thread = (voxel, 8-channel group).
__global__ void warpvar_kernel(const float* __restrict__ feat,
                               const float* __restrict__ dv,
                               const float* __restrict__ rt,
                               _Float16* __restrict__ var)
{
    int tid = blockIdx.x * blockDim.x + threadIdx.x;
    if (tid >= NVOX * 4) return;
    int cg = tid & 3;
    int vox = tid >> 2;
    int w = vox % WW, h = (vox / WW) % HH;
    float depth = dv[vox];

    float s[8], q[8];
#pragma unroll
    for (int i = 0; i < 8; ++i) {
        float r = feat[(size_t)(cg * 8 + i) * HW + h * WW + w];
        s[i] = r; q[i] = r * r;
    }
    const float fx = (float)w, fy = (float)h;
    for (int v = 1; v < VV; ++v) {
        const float* R = rt + (v - 1) * 12;
        float X = (R[0] * fx + R[1] * fy + R[2]) * depth + R[9];
        float Y = (R[3] * fx + R[4] * fy + R[5]) * depth + R[10];
        float Z = (R[6] * fx + R[7] * fy + R[8]) * depth + R[11];
        float iz = 1.f / Z;
        float px = X * iz, py = Y * iz;   // grid-normalize/denormalize cancels
        float x0f = floorf(px), y0f = floorf(py);
        float wx1 = px - x0f, wx0 = 1.f - wx1;
        float wy1 = py - y0f, wy0 = 1.f - wy1;
        int x0 = (int)x0f, y0 = (int)y0f;
        const float* sf = feat + (size_t)v * CC * HW;
        float acc[8];
#pragma unroll
        for (int i = 0; i < 8; ++i) acc[i] = 0.f;
#pragma unroll
        for (int k = 0; k < 4; ++k) {
            int xi = x0 + (k & 1), yi = y0 + (k >> 1);
            float wk = ((k & 1) ? wx1 : wx0) * ((k >> 1) ? wy1 : wy0);
            bool valid = (xi >= 0) && (xi < WW) && (yi >= 0) && (yi < HH);
            if (valid && wk != 0.f) {
                int base = yi * WW + xi;
#pragma unroll
                for (int i = 0; i < 8; ++i)
                    acc[i] += wk * sf[(size_t)(cg * 8 + i) * HW + base];
            }
        }
#pragma unroll
        for (int i = 0; i < 8; ++i) { s[i] += acc[i]; q[i] += acc[i] * acc[i]; }
    }
    V8H o;
    constexpr float inv3 = 1.f / 3.f;
#pragma unroll
    for (int i = 0; i < 8; ++i) {
        float m = s[i] * inv3;
        o.h[i] = (_Float16)(q[i] * inv3 - m * m);
    }
    *reinterpret_cast<f4raw*>(var + (size_t)vox * 32 + cg * 8) = o.q;
}

// conv0 (32ch -> 16pad, relu). One wave per 16-position tile; 27 WMMA.
// Weights staged once per block in LDS; A operands come from ds_load_b128.
// B lane layout: lane<16 -> N=lane, K=0..15; lane>=16 -> N=lane-16, K=16..31.
__global__ void conv0_wmma_kernel(const _Float16* __restrict__ var,
                                  const _Float16* __restrict__ w0f,
                                  const float* __restrict__ bfmt,
                                  _Float16* __restrict__ h1)
{
    __shared__ _Float16 sW[27 * 32 * 16];   // 27648 B
    __shared__ float    sB[16];

    const int tid = threadIdx.x;
    stage_to_lds(w0f, sW, 27 * 32 * 16, tid, blockDim.x);
    if (tid < 16) sB[tid] = bfmt[tid];
    __syncthreads();

    int wave = (blockIdx.x * blockDim.x + tid) >> 5;
    int lane = tid & 31;
    if (wave < TILES) {
        int wt = wave % WTILES;
        int hh = (wave / WTILES) % HH;
        int dd = wave / (WTILES * HH);
        int nIdx = lane & 15, khalf = lane >> 4;

        v8f c = {};
        const f4raw z0 = {0.f, 0.f, 0.f, 0.f};
#pragma unroll
        for (int kd = 0; kd < 3; ++kd) {
            int zd = dd + kd - 1;
            if ((unsigned)zd >= (unsigned)DD) continue;   // uniform skip == zero pad
#pragma unroll
            for (int kh = 0; kh < 3; ++kh) {
                int zh = hh + kh - 1;
                if ((unsigned)zh >= (unsigned)HH) continue;
                const _Float16* row = var + ((size_t)zd * HH + zh) * WW * 32;
                __builtin_prefetch(row + (size_t)WW * 32, 0, 1);   // next h row
#pragma unroll
                for (int kw = 0; kw < 3; ++kw) {
                    int tap = (kd * 3 + kh) * 3 + kw;
                    V16U a;
                    const f4raw* ap = reinterpret_cast<const f4raw*>(sW + ((tap * 32 + lane) << 4));
                    a.q[0] = ap[0]; a.q[1] = ap[1];      // ds_load_b128 x2
                    int pw = wt * 16 + nIdx + kw - 1;
                    bool inw = (unsigned)pw < (unsigned)WW;
                    int pwc = inw ? pw : (pw < 0 ? 0 : WW - 1);   // clamped, always valid
                    V16U b;
                    const f4raw* bp = reinterpret_cast<const f4raw*>(row + (size_t)pwc * 32 + khalf * 16);
                    b.q[0] = inw ? bp[0] : z0;           // SAME zero padding in W
                    b.q[1] = inw ? bp[1] : z0;
                    c = __builtin_amdgcn_wmma_f32_16x16x32_f16(
                            false, a.v, false, b.v, (short)0, c, false, false);
                }
            }
        }
        // C/D layout: lane holds column N=nIdx, VGPR j -> M = khalf*8 + j
        V8H o;
#pragma unroll
        for (int j = 0; j < 8; ++j) {
            float vv = c[j] + sB[khalf * 8 + j];
            o.h[j] = (_Float16)(vv > 0.f ? vv : 0.f);   // padded ch 8..15 -> exact 0
        }
        int wcol = wt * 16 + nIdx;
        *reinterpret_cast<f4raw*>(h1 + ((size_t)(dd * HH + hh) * WW + wcol) * 16 + khalf * 8) = o.q;
    }
}

// conv1 (mode 0: 16pad -> 16pad + relu) / conv2 (mode 1: 16pad -> 1 f32).
// K=32 spans two adjacent w-positions x 16 padded channels; 18 WMMA.
__global__ void convB_wmma_kernel(const _Float16* __restrict__ in,
                                  const _Float16* __restrict__ wf,
                                  const float* __restrict__ bfmt,
                                  _Float16* __restrict__ out16,
                                  float* __restrict__ outf,
                                  int mode)
{
    __shared__ _Float16 sW[9 * 2 * 32 * 16];   // 18432 B
    __shared__ float    sB[16];

    const int tid = threadIdx.x;
    stage_to_lds(wf, sW, 9 * 2 * 32 * 16, tid, blockDim.x);
    if (tid < 16) sB[tid] = bfmt[tid];
    __syncthreads();

    int wave = (blockIdx.x * blockDim.x + tid) >> 5;
    int lane = tid & 31;
    if (wave < TILES) {
        int wt = wave % WTILES;
        int hh = (wave / WTILES) % HH;
        int dd = wave / (WTILES * HH);
        int nIdx = lane & 15, khalf = lane >> 4;

        v8f c = {};
        const f4raw z0 = {0.f, 0.f, 0.f, 0.f};
#pragma unroll
        for (int r = 0; r < 9; ++r) {
            int kd = r / 3, kh = r % 3;
            int zd = dd + kd - 1;
            if ((unsigned)zd >= (unsigned)DD) continue;
            int zh = hh + kh - 1;
            if ((unsigned)zh >= (unsigned)HH) continue;
            const _Float16* row = in + ((size_t)zd * HH + zh) * WW * 16;
#pragma unroll
            for (int s = 0; s < 2; ++s) {
                V16U a;
                const f4raw* ap = reinterpret_cast<const f4raw*>(sW + (((r * 2 + s) * 32 + lane) << 4));
                a.q[0] = ap[0]; a.q[1] = ap[1];          // ds_load_b128 x2
                // lane's 16 halves = one position's 16 padded channels
                int pw = wt * 16 + nIdx + s * 2 + khalf - 1;
                bool inw = (unsigned)pw < (unsigned)WW;
                int pwc = inw ? pw : (pw < 0 ? 0 : WW - 1);
                V16U b;
                const f4raw* bp = reinterpret_cast<const f4raw*>(row + (size_t)pwc * 16);
                b.q[0] = inw ? bp[0] : z0;
                b.q[1] = inw ? bp[1] : z0;
                c = __builtin_amdgcn_wmma_f32_16x16x32_f16(
                        false, a.v, false, b.v, (short)0, c, false, false);
            }
        }
        if (mode == 0) {
            V8H o;
#pragma unroll
            for (int j = 0; j < 8; ++j) {
                float vv = c[j] + sB[khalf * 8 + j];
                o.h[j] = (_Float16)(vv > 0.f ? vv : 0.f);
            }
            int wcol = wt * 16 + nIdx;
            *reinterpret_cast<f4raw*>(out16 + ((size_t)(dd * HH + hh) * WW + wcol) * 16 + khalf * 8) = o.q;
        } else {
            if (lane < 16)   // M=0 row lives in lanes 0..15, VGPR 0
                outf[(size_t)(dd * HH + hh) * WW + wt * 16 + nIdx] = c[0] + sB[0];
        }
    }
}

// Softmax over D, expected depth, 4-tap confidence. 3 passes over L2-resident cost.
__global__ void depth_conf_kernel(const float* __restrict__ cost,
                                  const float* __restrict__ dv,
                                  float* __restrict__ outp)
{
    int pix = blockIdx.x * blockDim.x + threadIdx.x;
    if (pix >= HW) return;
    float m = -3.4e38f;
    for (int d = 0; d < DD; ++d) m = fmaxf(m, cost[(size_t)d * HW + pix]);
    float Z = 0.f, dAcc = 0.f, iAcc = 0.f;
    for (int d = 0; d < DD; ++d) {
        float e = __expf(cost[(size_t)d * HW + pix] - m);
        Z += e;
        dAcc += e * dv[(size_t)d * HW + pix];
        iAcc += e * (float)d;
    }
    float invZ = 1.f / Z;
    float depth = dAcc * invZ;
    int didx = (int)(iAcc * invZ);
    didx = didx < 0 ? 0 : (didx > DD - 1 ? DD - 1 : didx);
    float conf = 0.f;
    for (int d = didx - 1; d <= didx + 2; ++d)
        if (d >= 0 && d < DD) conf += __expf(cost[(size_t)d * HW + pix] - m);
    conf *= invZ;
    outp[pix] = depth;
    outp[HW + pix] = conf;
}

extern "C" void kernel_launch(void* const* d_in, const int* in_sizes, int n_in,
                              void* d_out, int out_size, void* d_ws, size_t ws_size,
                              hipStream_t stream)
{
    (void)in_sizes; (void)n_in; (void)out_size; (void)ws_size;
    const float* feat = (const float*)d_in[0];
    const float* proj = (const float*)d_in[1];
    const float* dv   = (const float*)d_in[2];
    const float* w0   = (const float*)d_in[3];
    const float* b0   = (const float*)d_in[4];
    const float* w1   = (const float*)d_in[5];
    const float* b1   = (const float*)d_in[6];
    const float* w2   = (const float*)d_in[7];
    const float* b2   = (const float*)d_in[8];

    char* ws = (char*)d_ws;
    float*    rt   = (float*)(ws + OFF_RT);
    float*    bfmt = (float*)(ws + OFF_BF);
    _Float16* w0f  = (_Float16*)(ws + OFF_W0F);
    _Float16* w1f  = (_Float16*)(ws + OFF_W1F);
    _Float16* w2f  = (_Float16*)(ws + OFF_W2F);
    _Float16* var  = (_Float16*)(ws + OFF_VAR);
    _Float16* h1   = (_Float16*)(ws + OFF_H1);
    _Float16* h2   = (_Float16*)(ws + OFF_H2);
    float*    cost = (float*)(ws + OFF_COST);   // aliases var (dead by then)

    prep_kernel<<<1, 256, 0, stream>>>(proj, w0, b0, w1, b1, w2, b2,
                                       rt, bfmt, w0f, w1f, w2f);
    warpvar_kernel<<<(NVOX * 4 + 255) / 256, 256, 0, stream>>>(feat, dv, rt, var);
    conv0_wmma_kernel<<<(TILES + 7) / 8, 256, 0, stream>>>(var, w0f, bfmt, h1);
    convB_wmma_kernel<<<(TILES + 7) / 8, 256, 0, stream>>>(h1, w1f, bfmt + 16, h2, nullptr, 0);
    convB_wmma_kernel<<<(TILES + 7) / 8, 256, 0, stream>>>(h2, w2f, bfmt + 32, nullptr, cost, 1);
    depth_conf_kernel<<<(HW + 255) / 256, 256, 0, stream>>>(cost, dv, (float*)d_out);
}
